// MoELayer_13383118094835
// MI455X (gfx1250) — matrix operand order repaired and checked by reading, not compiled, not run on hardware
//
#include <hip/hip_runtime.h>

typedef __attribute__((ext_vector_type(8)))  float  v8f;
typedef __attribute__((ext_vector_type(8)))  __bf16 v8bf;
typedef __attribute__((ext_vector_type(16))) __bf16 v16bf;

#define HIDDEN   1024
#define EXPERTS  8
#define FDIM     4096
#define TOKENS   8192
#define TILE_M   128
#define PADT     (TOKENS + EXPERTS * TILE_M)   // 9216 padded rows
#define MAXTILES 80
#define KSTEP    64   // LDS stage depth: two WMMA K=32 sub-steps per barrier pair
#define LDA      72   // LDS row stride (elements); 144B = 9*16B: b128-aligned, banks spread

// meta layout (ints): [0]=n_tiles, [1..8]=padded start per expert,
// [16..95]=tile_expert, [96..175]=tile_row, [200..207]=counts, [208..215]=cursor

union Frag16 { v16bf v; v8bf h[2]; };

__device__ inline v8bf cvt8(float4 a, float4 b) {
    v8bf o;
    o[0] = (__bf16)a.x; o[1] = (__bf16)a.y; o[2] = (__bf16)a.z; o[3] = (__bf16)a.w;
    o[4] = (__bf16)b.x; o[5] = (__bf16)b.y; o[6] = (__bf16)b.z; o[7] = (__bf16)b.w;
    return o;
}

__global__ void k_init(int* __restrict__ perm, int* __restrict__ meta) {
    int i = blockIdx.x * blockDim.x + threadIdx.x;
    if (i < PADT) perm[i] = -1;
    if (i < 16) meta[200 + i] = 0;   // counts + cursors
}

__global__ __launch_bounds__(256) void k_gate(const float* __restrict__ x,
                                              const float* __restrict__ gw,
                                              const float* __restrict__ gb,
                                              int* __restrict__ eidx,
                                              int* __restrict__ meta) {
    int wave = threadIdx.x >> 5, lane = threadIdx.x & 31;
    int tok = blockIdx.x * 8 + wave;
    float a[8];
#pragma unroll
    for (int e = 0; e < 8; e++) a[e] = 0.f;
    const float* xp = x + (size_t)tok * HIDDEN;
    for (int h = lane; h < HIDDEN; h += 32) {
        float xv = xp[h];
        const float4* g = (const float4*)(gw + (size_t)h * EXPERTS);
        float4 g0 = g[0], g1 = g[1];
        a[0] += xv * g0.x; a[1] += xv * g0.y; a[2] += xv * g0.z; a[3] += xv * g0.w;
        a[4] += xv * g1.x; a[5] += xv * g1.y; a[6] += xv * g1.z; a[7] += xv * g1.w;
    }
#pragma unroll
    for (int e = 0; e < 8; e++)
#pragma unroll
        for (int off = 16; off > 0; off >>= 1) a[e] += __shfl_xor(a[e], off, 32);
    if (lane == 0) {
        int best = 0; float bv = a[0] + gb[0];
#pragma unroll
        for (int e = 1; e < 8; e++) {
            float v = a[e] + gb[e];
            if (v > bv) { bv = v; best = e; }   // strict > keeps first tie = jnp.argmax
        }
        eidx[tok] = best;
        atomicAdd(&meta[200 + best], 1);
    }
}

__global__ void k_scan(int* __restrict__ meta) {
    if (threadIdx.x != 0 || blockIdx.x != 0) return;
    int P = 0, t = 0;
    for (int e = 0; e < 8; e++) {
        meta[1 + e] = P;
        int nt = (meta[200 + e] + TILE_M - 1) / TILE_M;
        for (int j = 0; j < nt; j++) { meta[16 + t] = e; meta[96 + t] = P + j * TILE_M; t++; }
        P += nt * TILE_M;
    }
    meta[0] = t;   // <= 71
}

__global__ void k_scatter(const int* __restrict__ eidx, int* __restrict__ meta,
                          int* __restrict__ perm) {
    int i = blockIdx.x * blockDim.x + threadIdx.x;
    if (i >= TOKENS) return;
    int e = eidx[i];
    int r = atomicAdd(&meta[208 + e], 1);
    perm[meta[1 + e] + r] = i;
}

// Tiled GEMM: 128(M) x 128(N) per block, K-stage 64 (2 WMMA sub-steps / barrier pair),
// 8 waves each owning 32x64 output -> 16 WMMAs per stage per wave.
// PASS1: A = gather(x) f32->bf16, B = w1[e], epilogue relu(+b1) -> bf16 hbuf (sorted rows)
// PASS2: A = hbuf (contiguous bf16), B = w2[e], epilogue (+b2) f32 scatter to out[perm[row]]
template <int PASS1>
__global__ __launch_bounds__(256) void k_ffn(const float* __restrict__ Xf,
                                             const __bf16* __restrict__ Hb,
                                             const float* __restrict__ W,
                                             const float* __restrict__ Bias,
                                             const int* __restrict__ meta,
                                             const int* __restrict__ perm,
                                             __bf16* __restrict__ hout,
                                             float* __restrict__ yout) {
    constexpr int K  = PASS1 ? HIDDEN : FDIM;
    constexpr int NT = PASS1 ? FDIM : HIDDEN;

    __shared__ __align__(16) __bf16 lA[TILE_M * LDA];
    __shared__ __align__(16) __bf16 lB[TILE_M * LDA];
    __shared__ int s_tok[TILE_M];

    int tid = threadIdx.x;
    if ((int)blockIdx.x >= meta[0]) return;               // uniform exit
    int e    = meta[16 + blockIdx.x];
    int row0 = meta[96 + blockIdx.x];
    int col0 = (int)blockIdx.y * 128;
    if (tid < TILE_M) s_tok[tid] = perm[row0 + tid];
    __syncthreads();

    v8f acc[2][4] = {};
    const float* Wb = W + (size_t)e * K * NT + col0;

    for (int kt = 0; kt < K; kt += KSTEP) {
        // ---- A tile: 128 rows x 64 k -> lA[row][k]; 2 threads/row, 32 k each ----
        {
            int row = tid >> 1, koff = (tid & 1) * 32;
            if (PASS1) {
                int tok = s_tok[row];
                float4 f[8];
#pragma unroll
                for (int i = 0; i < 8; i++) f[i] = (float4){0.f, 0.f, 0.f, 0.f};
                if (tok >= 0) {
                    const float4* p = (const float4*)(Xf + (size_t)tok * HIDDEN + kt + koff);
#pragma unroll
                    for (int i = 0; i < 8; i++) f[i] = p[i];
                }
#pragma unroll
                for (int i = 0; i < 4; i++)
                    *(v8bf*)&lA[row * LDA + koff + 8 * i] = cvt8(f[2 * i], f[2 * i + 1]);
            } else {
                const v8bf* p = (const v8bf*)(Hb + (size_t)(row0 + row) * FDIM + kt + koff);
#pragma unroll
                for (int i = 0; i < 4; i++)
                    *(v8bf*)&lA[row * LDA + koff + 8 * i] = p[i];
            }
        }
        // ---- B tile: rows kt..kt+63 of W[e], cols col0..+127, transposed -> lB[n][k] ----
        {
            int kp = tid >> 4;              // 0..15 -> k pair base
            int n0 = (tid & 15) * 8;        // 0..120
#pragma unroll
            for (int kk = 0; kk < KSTEP; kk += 32) {
                const float* w0 = Wb + (size_t)(kt + kk + 2 * kp) * NT + n0;
                float4 a0 = ((const float4*)w0)[0], a1 = ((const float4*)w0)[1];
                float4 b0 = ((const float4*)(w0 + NT))[0], b1 = ((const float4*)(w0 + NT))[1];
                float r0[8] = {a0.x, a0.y, a0.z, a0.w, a1.x, a1.y, a1.z, a1.w};
                float r1[8] = {b0.x, b0.y, b0.z, b0.w, b1.x, b1.y, b1.z, b1.w};
#pragma unroll
                for (int i = 0; i < 8; i++) {
                    union { __bf16 b[2]; unsigned u; } pk;
                    pk.b[0] = (__bf16)r0[i]; pk.b[1] = (__bf16)r1[i];
                    *(unsigned*)&lB[(n0 + i) * LDA + kk + 2 * kp] = pk.u;
                }
            }
            if (kt + KSTEP < K) __builtin_prefetch(Wb + (size_t)(kt + KSTEP + 2 * kp) * NT + n0, 0, 3);
        }
        __syncthreads();
        // ---- 16 WMMAs per wave (2 K-sub-steps x 2 M-frags x 4 N-frags), EXEC all ones ----
        {
            int lane = tid & 31, w = tid >> 5;
            int wm = w & 3, wn = w >> 2;
            int half = lane >> 4, l16 = lane & 15;
#pragma unroll
            for (int ks = 0; ks < KSTEP; ks += 32) {
                Frag16 a0, a1, b;
                const __bf16* pa = &lA[(wm * 32 + l16) * LDA + ks];
                a0.h[0] = *(const v8bf*)(pa + half * 8);        // K = halfsel*8 + 0..7
                a0.h[1] = *(const v8bf*)(pa + 16 + half * 8);   // K = 16 + halfsel*8 + 0..7
                pa += 16 * LDA;
                a1.h[0] = *(const v8bf*)(pa + half * 8);
                a1.h[1] = *(const v8bf*)(pa + 16 + half * 8);
#pragma unroll
                for (int fj = 0; fj < 4; fj++) {
                    const __bf16* pb = &lB[(wn * 64 + fj * 16 + l16) * LDA + ks + half * 8];
                    b.h[0] = *(const v8bf*)pb;                  // K = halfsel*8 + 0..7, 16 + halfsel*8 + 0..7
                    b.h[1] = *(const v8bf*)(pb + 16);
                    acc[0][fj] = __builtin_amdgcn_wmma_f32_16x16x32_bf16(
                        false, a0.v, false, b.v, (short)0, acc[0][fj], false, false);
                    acc[1][fj] = __builtin_amdgcn_wmma_f32_16x16x32_bf16(
                        false, a1.v, false, b.v, (short)0, acc[1][fj], false, false);
                }
            }
        }
        __syncthreads();
    }
    // ---- epilogue ----
    {
        int lane = tid & 31, w = tid >> 5;
        int wm = w & 3, wn = w >> 2;
        int half = lane >> 4, l16 = lane & 15;
#pragma unroll
        for (int fi = 0; fi < 2; fi++)
#pragma unroll
            for (int fj = 0; fj < 4; fj++) {
                int colf = col0 + wn * 64 + fj * 16 + l16;
                float bias = Bias[(size_t)e * NT + colf];
#pragma unroll
                for (int r = 0; r < 8; r++) {
                    int rowf = wm * 32 + fi * 16 + 8 * half + r;  // C/D: VGPR r = row r + 8*halfsel
                    float v = acc[fi][fj][r] + bias;
                    if (PASS1) {
                        v = v > 0.f ? v : 0.f;
                        hout[(size_t)(row0 + rowf) * FDIM + colf] = (__bf16)v;
                    } else {
                        int tok = s_tok[rowf];
                        if (tok >= 0) yout[(size_t)tok * HIDDEN + colf] = v;
                    }
                }
            }
    }
}

extern "C" void kernel_launch(void* const* d_in, const int* in_sizes, int n_in,
                              void* d_out, int out_size, void* d_ws, size_t ws_size,
                              hipStream_t stream) {
    (void)in_sizes; (void)n_in; (void)out_size; (void)ws_size;
    const float* x  = (const float*)d_in[0];
    const float* gw = (const float*)d_in[1];
    const float* gb = (const float*)d_in[2];
    const float* w1 = (const float*)d_in[3];
    const float* b1 = (const float*)d_in[4];
    const float* w2 = (const float*)d_in[5];
    const float* b2 = (const float*)d_in[6];
    float* out = (float*)d_out;

    char* ws = (char*)d_ws;
    int*    meta = (int*)ws;                       // 1 KB
    int*    eidx = (int*)(ws + 1024);              // 32 KB
    int*    perm = (int*)(ws + 1024 + 32768);      // 36 KB (PADT ints)
    __bf16* hbuf = (__bf16*)(ws + 71680);          // 9216*4096 bf16 = 75.5 MB (256B-aligned off)

    k_init<<<PADT / 256, 256, 0, stream>>>(perm, meta);
    k_gate<<<TOKENS / 8, 256, 0, stream>>>(x, gw, gb, eidx, meta);
    k_scan<<<1, 1, 0, stream>>>(meta);
    k_scatter<<<TOKENS / 256, 256, 0, stream>>>(eidx, meta, perm);
    k_ffn<1><<<dim3(MAXTILES, FDIM / 128), 256, 0, stream>>>(
        x, nullptr, w1, b1, meta, perm, hbuf, nullptr);
    k_ffn<0><<<dim3(MAXTILES, HIDDEN / 128), 256, 0, stream>>>(
        nullptr, hbuf, w2, b2, meta, perm, nullptr, out);
}
